// SoftmaxMatcherBlock_81174881894613
// MI455X (gfx1250) — matrix operand order repaired and checked by reading, not compiled, not run on hardware
//
#include <hip/hip_runtime.h>
#include <hip/hip_fp16.h>
#include <math.h>

typedef __attribute__((ext_vector_type(16))) _Float16 v16h;
typedef __attribute__((ext_vector_type(2)))  __fp16   f16x2;
typedef __attribute__((ext_vector_type(8)))  float    v8f;

#define BB 4
#define CC 64
#define NN 1024
#define HH 64
#define WW 1024
#define MM (HH*WW)        // 65536
#define QSCALE 2.2542110f // (1/(C*TEMP)) * log2(e): softmax in exp2 domain
#define VCH 80            // value channels padded (64 desc + 3 coord + 1 weight + 12 pad)
#define NT (NN/16)        // 64 16-query tiles / batch
#define QT (NN/32)        // 32 32-query wave tiles / batch
#define MT (MM/16)        // 4096 key tiles / batch
#define MC (MM/32)        // 2048 key chunks / batch
#define FRAG 512          // halves per fragment (32 lanes * 16)
#define NWAVE 4           // waves per block (key split inside block)
#define KSPLIT 2          // key split across blocks (combined in pass 2)
#define CPW (MC/(KSPLIT*NWAVE))  // 256 chunks per wave
#define PSTRIDE 70        // partial record: m, l, 68 channels

// ws layout (in _Float16 units)
#define WSQ_OFF 0
#define WSQ_SZ  (BB*NT*2*FRAG)            // 262144
#define WSK_OFF (WSQ_OFF+WSQ_SZ)
#define WSK_SZ  ((size_t)BB*MT*2*FRAG)    // 16777216
#define WSV_OFF (WSK_OFF+WSK_SZ)
#define WSV_SZ  ((size_t)BB*MC*5*FRAG)    // 20971520
#define WSP_OFF (WSV_OFF+WSV_SZ)          // float partials after this (halves idx)

// output offsets (float units)
#define OUT_COORD 0                        // (B,3,N)
#define OUT_W     12288                    // (B,1,N)
#define OUT_DESC  16384                    // (B,64,N)
#define OUT_2D    278528                   // (B,N,2)
#define OUT_VALID 286720                   // (B,1,N) int32

#if __has_builtin(__builtin_amdgcn_exp2f)
#define EXP2(x) __builtin_amdgcn_exp2f(x)
#else
#define EXP2(x) exp2f(x)
#endif

// single-VALU cross-half (xor 16) exchange
__device__ __forceinline__ float xhalf16(float x) {
#if __has_builtin(__builtin_amdgcn_permlanex16)
    int i = __builtin_bit_cast(int, x);
    i = __builtin_amdgcn_permlanex16(i, i, 0x76543210, 0xfedcba98, false, false);
    return __builtin_bit_cast(float, i);
#else
    return __shfl_xor(x, 16, 32);
#endif
}

// ---- A-fragment (16x32 f16) element index: lane l, elem e -> K index ----
__device__ __forceinline__ int a_frag_k(int l, int e) {
    int j = e >> 1, p = e & 1;
    return (j < 4) ? (((l < 16) ? 0 : 8) + 2 * j + p)
                   : (((l < 16) ? 16 : 24) + 2 * (j - 4) + p);
}
// key-octet permutation (swap octets 1 and 2), unused: V packing keeps a_frag_k order,
// so the S^T accumulator halves feed the PV B-operand with no lane exchange
__device__ __forceinline__ int kv_perm(int k) {
    return (k < 8) ? k : (k < 16) ? k + 8 : (k < 24) ? k - 8 : k;
}

// ============ pack Q: zn(src_desc)*QSCALE -> B-frag f16 (ch x queries) =======
__global__ __launch_bounds__(32) void pack_q(const float* __restrict__ src,
                                             _Float16* __restrict__ wsQ) {
    int tile = blockIdx.x, b = blockIdx.y, l = threadIdx.x;
    __shared__ float ld[16][CC];
    __shared__ float mu[16], rs[16];
    if (l < 16) {
        int n = tile * 16 + l;
        float s = 0.f, s2 = 0.f;
        for (int c = 0; c < CC; c++) {
            float v = src[(b * CC + c) * NN + n];
            ld[l][c] = v; s += v; s2 += v * v;
        }
        float m = s / CC;
        float var = (s2 - CC * m * m) / (CC - 1);
        mu[l] = m; rs[l] = rsqrtf(fmaxf(var, 1e-24f)) * QSCALE;
    }
    __syncthreads();
    int n16 = l & 15, kb = (l < 16) ? 0 : 16;
    for (int kt = 0; kt < 2; kt++) {
        v16h bf;
        #pragma unroll
        for (int e = 0; e < 16; e++) {
            int ch = kt * 32 + a_frag_k(l, e);
            bf[e] = (_Float16)((ld[n16][ch] - mu[n16]) * rs[n16]);
        }
        *(v16h*)&wsQ[((size_t)(b * NT + tile) * 2 + kt) * FRAG + l * 16] = bf;
    }
}

// ============ pack K: zn(tgt_desc) -> A-frag f16 (keys x channels) ===========
__global__ __launch_bounds__(32) void pack_k(const float* __restrict__ tgt,
                                             _Float16* __restrict__ wsK) {
    int tile = blockIdx.x, b = blockIdx.y, l = threadIdx.x;
    __shared__ float ld[16][CC];
    __shared__ float mu[16], rs[16];
    if (l < 16) {
        int m = tile * 16 + l;
        float s = 0.f, s2 = 0.f;
        for (int c = 0; c < CC; c++) {
            float v = tgt[(size_t)(b * CC + c) * MM + m];
            ld[l][c] = v; s += v; s2 += v * v;
        }
        float mean = s / CC;
        float var = (s2 - CC * mean * mean) / (CC - 1);
        mu[l] = mean; rs[l] = rsqrtf(fmaxf(var, 1e-24f));
    }
    __syncthreads();
    int k16 = l & 15;
    for (int ct = 0; ct < 2; ct++) {
        v16h af;
        #pragma unroll
        for (int e = 0; e < 16; e++) {
            int ch = ct * 32 + a_frag_k(l, e);
            af[e] = (_Float16)((ld[k16][ch] - mu[k16]) * rs[k16]);
        }
        *(v16h*)&wsK[((size_t)(b * MT + tile) * 2 + ct) * FRAG + l * 16] = af;
    }
}

// ==== pack V: [desc|coords|weights|pad] -> A-frag f16 (ch x keys) ===
__global__ __launch_bounds__(32) void pack_v(const float* __restrict__ tgtd,
                                             const float* __restrict__ tgtc,
                                             const float* __restrict__ tgtw,
                                             _Float16* __restrict__ wsV) {
    int chunk = blockIdx.x, b = blockIdx.y, l = threadIdx.x;
    for (int vt = 0; vt < 5; vt++) {
        int ch = vt * 16 + (l & 15);
        const float* p = nullptr;
        if (ch < 64)       p = tgtd + (size_t)(b * CC + ch) * MM + chunk * 32;
        else if (ch < 67)  p = tgtc + (size_t)(b * 3 + (ch - 64)) * MM + chunk * 32;
        else if (ch == 67) p = tgtw + (size_t)b * MM + chunk * 32;
        v16h out;
        if (p) {
            #pragma unroll
            for (int e = 0; e < 16; e++) out[e] = (_Float16)p[a_frag_k(l, e)];
        } else {
            #pragma unroll
            for (int e = 0; e < 16; e++) out[e] = (_Float16)0.f;
        }
        *(v16h*)&wsV[((size_t)(b * MC + chunk) * 5 + vt) * FRAG + l * 16] = out;
    }
}

// ===================== flash pass 1: partial attention ======================
// 32 queries per wave (2 groups), S^T = K*Q^T, per-lane softmax stats,
// O^T += V^T * P^T. Keys split over 4 waves * KSPLIT blocks.
__global__ __launch_bounds__(32*NWAVE) void flash(const _Float16* __restrict__ wsQ,
                                                  const _Float16* __restrict__ wsK,
                                                  const _Float16* __restrict__ wsV,
                                                  float* __restrict__ wsP) {
    int qt = blockIdx.x, b = blockIdx.y, ks = blockIdx.z;
    int wave = threadIdx.x >> 5, l = threadIdx.x & 31;
    __shared__ float obuf[NWAVE][32][VCH];
    __shared__ float mbuf[NWAVE][32];
    __shared__ float lbuf[NWAVE][32];

    v16h qb[2][2];
    #pragma unroll
    for (int g = 0; g < 2; g++)
        #pragma unroll
        for (int kt = 0; kt < 2; kt++)
            qb[g][kt] = *(const v16h*)&wsQ[((size_t)(b * NT + qt * 2 + g) * 2 + kt) * FRAG + l * 16];

    v8f z = {0.f, 0.f, 0.f, 0.f, 0.f, 0.f, 0.f, 0.f};
    v8f o[2][5];
    #pragma unroll
    for (int g = 0; g < 2; g++)
        #pragma unroll
        for (int vt = 0; vt < 5; vt++) o[g][vt] = z;
    float rowm[2] = {-3.0e38f, -3.0e38f};
    float rowl[2] = {0.f, 0.f};

    int it0 = ks * (MC / KSPLIT) + wave * CPW, it1 = it0 + CPW;
    for (int it = it0; it < it1; it++) {
        size_t kbase = ((size_t)(b * MT + it * 2)) * 2 * FRAG;
        v16h a00 = *(const v16h*)&wsK[kbase + 0 * FRAG + l * 16];
        v16h a01 = *(const v16h*)&wsK[kbase + 1 * FRAG + l * 16];
        v16h a10 = *(const v16h*)&wsK[kbase + 2 * FRAG + l * 16];
        v16h a11 = *(const v16h*)&wsK[kbase + 3 * FRAG + l * 16];
        size_t vbase = ((size_t)(b * MC + it) * 5) * FRAG + l * 16;
        v16h va[5];
        #pragma unroll
        for (int vt = 0; vt < 5; vt++) va[vt] = *(const v16h*)&wsV[vbase + vt * FRAG];
        if (it + 1 < it1) {  // global_prefetch_b8 next chunk
            __builtin_prefetch((const void*)&wsK[kbase + 4 * FRAG + l * 16], 0, 0);
            __builtin_prefetch((const void*)&wsV[vbase + 5 * FRAG], 0, 0);
        }
        #pragma unroll
        for (int g = 0; g < 2; g++) {
            // S^T: keys x queries, already scaled to exp2 domain via Q pack
            v8f s0 = __builtin_amdgcn_wmma_f32_16x16x32_f16(false, a00, false, qb[g][0], (short)0, z,  false, false);
            s0     = __builtin_amdgcn_wmma_f32_16x16x32_f16(false, a01, false, qb[g][1], (short)0, s0, false, false);
            v8f s1 = __builtin_amdgcn_wmma_f32_16x16x32_f16(false, a10, false, qb[g][0], (short)0, z,  false, false);
            s1     = __builtin_amdgcn_wmma_f32_16x16x32_f16(false, a11, false, qb[g][1], (short)0, s1, false, false);

            float mx = fmaxf(s0[0], s1[0]);
            #pragma unroll
            for (int r = 1; r < 8; r++) mx = fmaxf(mx, fmaxf(s0[r], s1[r]));
            mx = fmaxf(mx, xhalf16(mx));
            float nm = fmaxf(rowm[g], mx);
            float prev = rowm[g];
            rowm[g] = nm;
            if (__any(nm > prev)) {          // rare after warm-up: skip rescale
                float alpha = EXP2(prev - nm);
                rowl[g] *= alpha;
                #pragma unroll
                for (int vt = 0; vt < 5; vt++)
                    #pragma unroll
                    for (int r = 0; r < 8; r++) o[g][vt][r] *= alpha;
            }
            float p0[8], p1[8], ps = 0.f;
            #pragma unroll
            for (int r = 0; r < 8; r++) {
                p0[r] = EXP2(s0[r] - nm);
                p1[r] = EXP2(s1[r] - nm);
                ps += p0[r] + p1[r];
            }
            rowl[g] += ps + xhalf16(ps);
            // P^T B-fragment needs NO exchange: V was packed in a_frag_k key order,
            // matching the S^T accumulator rows, so pb = [p0[0..7], p1[0..7]] on all lanes.
            union { v16h v; f16x2 h2[8]; } u;
            #pragma unroll
            for (int j = 0; j < 4; j++) {
#if __has_builtin(__builtin_amdgcn_cvt_pkrtz)
                u.h2[j]     = __builtin_amdgcn_cvt_pkrtz(p0[2 * j], p0[2 * j + 1]);
                u.h2[4 + j] = __builtin_amdgcn_cvt_pkrtz(p1[2 * j], p1[2 * j + 1]);
#else
                u.h2[j][0] = (__fp16)p0[2 * j];     u.h2[j][1] = (__fp16)p0[2 * j + 1];
                u.h2[4 + j][0] = (__fp16)p1[2 * j]; u.h2[4 + j][1] = (__fp16)p1[2 * j + 1];
#endif
            }
            v16h pb = u.v;
            #pragma unroll
            for (int vt = 0; vt < 5; vt++)
                o[g][vt] = __builtin_amdgcn_wmma_f32_16x16x32_f16(false, va[vt], false, pb, (short)0, o[g][vt], false, false);
        }
    }

    // stage per-wave partials: O^T acc rows = channels, cols = queries
    int col = l & 15;
    bool lo = (l < 16);
    #pragma unroll
    for (int g = 0; g < 2; g++) {
        int q = g * 16 + col;
        #pragma unroll
        for (int r = 0; r < 8; r++) {
            int row = lo ? r : 8 + r;
            #pragma unroll
            for (int vt = 0; vt < 5; vt++) obuf[wave][q][vt * 16 + row] = o[g][vt][r];
        }
        if (lo) { mbuf[wave][q] = rowm[g]; lbuf[wave][q] = rowl[g]; }
    }
    __syncthreads();

    // intra-block flash-combine over the 4 waves -> one partial per key half
    if (threadIdx.x < 32) {
        int q = threadIdx.x;
        float M = mbuf[0][q];
        #pragma unroll
        for (int w = 1; w < NWAVE; w++) M = fmaxf(M, mbuf[w][q]);
        float wt[NWAVE], L = 0.f;
        #pragma unroll
        for (int w = 0; w < NWAVE; w++) {
            wt[w] = EXP2(mbuf[w][q] - M);
            L += lbuf[w][q] * wt[w];
        }
        float* P = wsP + (((size_t)(b * QT + qt) * KSPLIT + ks) * 32 + q) * PSTRIDE;
        P[0] = M; P[1] = L;
        for (int c = 0; c < 68; c++) {
            float v = 0.f;
            #pragma unroll
            for (int w = 0; w < NWAVE; w++) v += obuf[w][q][c] * wt[w];
            P[2 + c] = v;
        }
    }
}

// ============== pass 2: merge key halves + normalize + outputs ==============
__global__ __launch_bounds__(32) void combine(const float* __restrict__ wsP,
                                              float* __restrict__ out) {
    int qt = blockIdx.x, b = blockIdx.y, q = threadIdx.x;
    int n = qt * 32 + q;
    const float* P0 = wsP + (((size_t)(b * QT + qt) * KSPLIT + 0) * 32 + q) * PSTRIDE;
    const float* P1 = wsP + (((size_t)(b * QT + qt) * KSPLIT + 1) * 32 + q) * PSTRIDE;
    float M = fmaxf(P0[0], P1[0]);
    float w0 = EXP2(P0[0] - M), w1 = EXP2(P1[0] - M);
    float L = P0[1] * w0 + P1[1] * w1;
    float inv = 1.f / L;

    // z-normalize pseudo descriptors (ddof=1)
    float d[64], s = 0.f, s2 = 0.f;
    #pragma unroll
    for (int c = 0; c < 64; c++) {
        float v = (P0[2 + c] * w0 + P1[2 + c] * w1) * inv;
        d[c] = v; s += v; s2 += v * v;
    }
    float mean = s / 64.f;
    float var = (s2 - 64.f * mean * mean) / 63.f;
    float rstd = rsqrtf(fmaxf(var, 1e-24f));
    #pragma unroll
    for (int c = 0; c < 64; c++)
        out[OUT_DESC + (b * 64 + c) * NN + n] = (d[c] - mean) * rstd;

    float x  = (P0[66] * w0 + P1[66] * w1) * inv;   // ch 64
    float y  = (P0[67] * w0 + P1[67] * w1) * inv;   // ch 65
    float zc = (P0[68] * w0 + P1[68] * w1) * inv;   // ch 66
    float wq = (P0[69] * w0 + P1[69] * w1) * inv;   // ch 67
    out[OUT_COORD + (b * 3 + 0) * NN + n] = x;
    out[OUT_COORD + (b * 3 + 1) * NN + n] = y;
    out[OUT_COORD + (b * 3 + 2) * NN + n] = zc;
    out[OUT_W + b * NN + n] = wq;

    float r3 = sqrtf(x * x + y * y + zc * zc);
    float az = atan2f(y, x);
    float el = asinf(zc / (r3 + 1e-12f));
    const float PI = 3.14159265358979323846f;
    float fov_up = 3.0f * PI / 180.f;
    float fov = 28.0f * PI / 180.f;
    float u = 0.5f * (1.f - az / PI) * (float)WW;
    float vv = (1.f - (el + (fov - fov_up)) / fov) * (float)HH;
    u = fminf(fmaxf(u, 0.f), (float)WW - 1.f);
    vv = fminf(fmaxf(vv, 0.f), (float)HH - 1.f);
    out[OUT_2D + (b * NN + n) * 2 + 0] = u;
    out[OUT_2D + (b * NN + n) * 2 + 1] = vv;
    ((int*)out)[OUT_VALID + b * NN + n] = 1;
}

extern "C" void kernel_launch(void* const* d_in, const int* in_sizes, int n_in,
                              void* d_out, int out_size, void* d_ws, size_t ws_size,
                              hipStream_t stream) {
    (void)in_sizes; (void)n_in; (void)out_size; (void)ws_size;
    // inputs: src_coords (unused), tgt_coords, tgt_weights, src_desc, tgt_desc
    const float* tgt_coords  = (const float*)d_in[1];
    const float* tgt_weights = (const float*)d_in[2];
    const float* src_desc    = (const float*)d_in[3];
    const float* tgt_desc    = (const float*)d_in[4];
    float* out = (float*)d_out;
    _Float16* ws = (_Float16*)d_ws;
    _Float16* wsQ = ws + WSQ_OFF;
    _Float16* wsK = ws + WSK_OFF;
    _Float16* wsV = ws + WSV_OFF;
    float* wsP = (float*)(ws + WSP_OFF);

    pack_q<<<dim3(NT, BB), 32, 0, stream>>>(src_desc, wsQ);
    pack_k<<<dim3(MT, BB), 32, 0, stream>>>(tgt_desc, wsK);
    pack_v<<<dim3(MC, BB), 32, 0, stream>>>(tgt_desc, tgt_coords, tgt_weights, wsV);
    flash<<<dim3(QT, BB, KSPLIT), 32 * NWAVE, 0, stream>>>(wsQ, wsK, wsV, wsP);
    combine<<<dim3(QT, BB), 32, 0, stream>>>(wsP, out);
}
